// OmniMoE_Pytorch_83150566850563
// MI455X (gfx1250) — matrix operand order repaired and checked by reading, not compiled, not run on hardware
//
#include <hip/hip_runtime.h>
#include <hip/hip_bf16.h>
#include <math.h>

// ---------------------------------------------------------------------------
// Types for WMMA (gfx1250, wave32)
// ---------------------------------------------------------------------------
typedef __attribute__((ext_vector_type(16))) __bf16 v16bf;
typedef __attribute__((ext_vector_type(8)))  float  v8f;

union FragBF {
    uint4        q[2];
    unsigned int u[8];
    v16bf        v;
};

#define NEG_INF (-1e30f)

__device__ __forceinline__ unsigned short f2bf(float f) {
    unsigned int u = __float_as_uint(f);
    unsigned int r = u + 0x7FFFu + ((u >> 16) & 1u);
    return (unsigned short)(r >> 16);
}

__device__ __forceinline__ float silu_f(float x) {
    return x / (1.f + __expf(-x));
}

// gfx1250 async copy: global -> LDS, 16 bytes per lane, tracked by ASYNCcnt
__device__ __forceinline__ void async_ld128(unsigned lds_off, const unsigned int* gaddr) {
    asm volatile("global_load_async_to_lds_b128 %0, %1, off"
                 :: "v"(lds_off), "v"(gaddr) : "memory");
}
__device__ __forceinline__ void wait_async0() {
    asm volatile("s_wait_asynccnt 0x0" ::: "memory");
}

// ---------------------------------------------------------------------------
// f32 -> bf16 conversion (grid-stride)
// ---------------------------------------------------------------------------
__global__ __launch_bounds__(256)
void cvt_bf16_kernel(const float* __restrict__ in, unsigned short* __restrict__ out, int n) {
    int i = blockIdx.x * 256 + threadIdx.x;
    int stride = gridDim.x * 256;
    for (; i < n; i += stride) out[i] = f2bf(in[i]);
}

// ---------------------------------------------------------------------------
// Tiled bf16 WMMA GEMM:  C[M,N] = A[M,K] @ W[N,K]^T
//   block tile 64(M) x 128(N) x 32(K); 8 waves, each wave 32x32 (2x2 wmma)
//   double-buffered LDS filled with global_load_async_to_lds_b128
// EPI 0: store f32                          (router logits)
// EPI 1: dual weights, h = silu(g)*u -> bf16 (gate/up fused)
// EPI 2: store f32 + addsrc                 (down proj + expert residual)
// ---------------------------------------------------------------------------
template <int EPI>
__global__ __launch_bounds__(256)
void gemm_bf16_kernel(const unsigned short* __restrict__ A,
                      const unsigned short* __restrict__ W0,
                      const unsigned short* __restrict__ W1,
                      const float* __restrict__ addsrc,
                      float* __restrict__ Cf,
                      unsigned short* __restrict__ Cb,
                      int M, int N, int K) {
    // row stride 20 dwords (80B): keeps every 4-dword fragment run 16B-aligned
    __shared__ __align__(16) unsigned int As[2][64][20];
    __shared__ __align__(16) unsigned int Bs0[2][128][20];
    __shared__ __align__(16) unsigned int Bs1[(EPI == 1) ? 2 : 1][(EPI == 1) ? 128 : 1][20];

    const int tid  = threadIdx.x;
    const int lane = tid & 31;
    const int wv   = tid >> 5;
    const int waveM = wv & 1;        // 2 waves along M
    const int waveN = wv >> 1;       // 4 waves along N
    const int bm = blockIdx.x * 64;
    const int bn = blockIdx.y * 128;

    v8f acc0[2][2];
    v8f acc1[2][2];
#pragma unroll
    for (int a = 0; a < 2; ++a)
#pragma unroll
        for (int b = 0; b < 2; ++b) {
            acc0[a][b] = v8f{0.f, 0.f, 0.f, 0.f, 0.f, 0.f, 0.f, 0.f};
            if (EPI == 1) acc1[a][b] = v8f{0.f, 0.f, 0.f, 0.f, 0.f, 0.f, 0.f, 0.f};
        }

    const unsigned int* Ad  = (const unsigned int*)A;
    const unsigned int* W0d = (const unsigned int*)W0;
    const unsigned int* W1d = (const unsigned int*)W1;
    const int Kd = K >> 1;           // dwords per row

    // tile-fill assignment: each thread owns one 16B chunk of A, two of each W
    const int rT = tid >> 2;         // 0..63
    const int cT = (tid & 3) * 4;    // dword column {0,4,8,12}
    const unsigned int* gA  = Ad  + (size_t)(bm + rT) * Kd + cT;
    const unsigned int* gB0 = W0d + (size_t)(bn + rT) * Kd + cT;
    const unsigned int* gB1 = W0d + (size_t)(bn + rT + 64) * Kd + cT;
    const unsigned int* gC0 = (EPI == 1) ? (W1d + (size_t)(bn + rT) * Kd + cT) : nullptr;
    const unsigned int* gC1 = (EPI == 1) ? (W1d + (size_t)(bn + rT + 64) * Kd + cT) : nullptr;

    const unsigned lA  = (unsigned)(size_t)&As[0][rT][cT];
    const unsigned lB0 = (unsigned)(size_t)&Bs0[0][rT][cT];
    const unsigned lB1 = (unsigned)(size_t)&Bs0[0][rT + 64][cT];
    const unsigned sA  = (unsigned)sizeof(As[0]);
    const unsigned sB  = (unsigned)sizeof(Bs0[0]);
    unsigned lC0 = 0, lC1 = 0, sC = 0;
    if (EPI == 1) {
        lC0 = (unsigned)(size_t)&Bs1[0][rT][cT];
        lC1 = (unsigned)(size_t)&Bs1[0][rT + 64][cT];
        sC  = (unsigned)sizeof(Bs1[0]);
    }

    const int g  = lane >> 4;        // lane half
    const int ln = lane & 15;
    const int nK = K >> 5;

    // prologue: async-fill buffer 0 with K-tile 0
    {
        async_ld128(lA, gA);
        async_ld128(lB0, gB0);
        async_ld128(lB1, gB1);
        if (EPI == 1) { async_ld128(lC0, gC0); async_ld128(lC1, gC1); }
    }

    for (int ki = 0; ki < nK; ++ki) {
        wait_async0();
        __syncthreads();
        const int buf = ki & 1;
        // issue async fill of the other buffer with the next K-tile
        if (ki + 1 < nK) {
            const int nb = buf ^ 1;
            const int kd = (ki + 1) << 4;   // dword offset of next K-tile
            async_ld128(lA + nb * sA, gA + kd);
            async_ld128(lB0 + nb * sB, gB0 + kd);
            async_ld128(lB1 + nb * sB, gB1 + kd);
            if (EPI == 1) {
                async_ld128(lC0 + nb * sC, gC0 + kd);
                async_ld128(lC1 + nb * sC, gC1 + kd);
            }
        }

        // ---- A fragments: 16x32 bf16, k = 16*g + s contiguous (b128 LDS reads) ----
        FragBF af[2];
#pragma unroll
        for (int mt = 0; mt < 2; ++mt) {
            int m = waveM * 32 + mt * 16 + ln;
            af[mt].q[0] = *(const uint4*)&As[buf][m][g * 8];
            af[mt].q[1] = *(const uint4*)&As[buf][m][g * 8 + 4];
        }
        // ---- B fragments: 32x16 bf16 (K pairs contiguous per W row) ----
        FragBF bf0[2], bf1[2];
#pragma unroll
        for (int nt = 0; nt < 2; ++nt) {
            int n = waveN * 32 + nt * 16 + ln;
            bf0[nt].q[0] = *(const uint4*)&Bs0[buf][n][g * 8];
            bf0[nt].q[1] = *(const uint4*)&Bs0[buf][n][g * 8 + 4];
            if (EPI == 1) {
                bf1[nt].q[0] = *(const uint4*)&Bs1[buf][n][g * 8];
                bf1[nt].q[1] = *(const uint4*)&Bs1[buf][n][g * 8 + 4];
            }
        }
        // ---- WMMA ----
#pragma unroll
        for (int mt = 0; mt < 2; ++mt)
#pragma unroll
            for (int nt = 0; nt < 2; ++nt) {
                acc0[mt][nt] = __builtin_amdgcn_wmma_f32_16x16x32_bf16(
                    false, af[mt].v, false, bf0[nt].v, (short)0, acc0[mt][nt], false, false);
                if (EPI == 1)
                    acc1[mt][nt] = __builtin_amdgcn_wmma_f32_16x16x32_bf16(
                        false, af[mt].v, false, bf1[nt].v, (short)0, acc1[mt][nt], false, false);
            }
        __syncthreads();
    }

    // ---- epilogue: C layout lane ln = N, vgpr r -> M = r + 8*g ----
#pragma unroll
    for (int mt = 0; mt < 2; ++mt) {
#pragma unroll
        for (int nt = 0; nt < 2; ++nt) {
            int mbase = bm + waveM * 32 + mt * 16 + g * 8;
            int n     = bn + waveN * 32 + nt * 16 + ln;
#pragma unroll
            for (int r = 0; r < 8; ++r) {
                int m = mbase + r;
                float v = acc0[mt][nt][r];
                if (EPI == 0) {
                    Cf[(size_t)m * N + n] = v;
                } else if (EPI == 1) {
                    float u = acc1[mt][nt][r];
                    Cb[(size_t)m * N + n] = f2bf(silu_f(v) * u);
                } else {
                    Cf[(size_t)m * N + n] = v + addsrc[(size_t)m * N + n];
                }
            }
        }
    }
}

// ---------------------------------------------------------------------------
// Per-feature (column) batch stats over logits [Ntok, 128] -> mean, rstd
// ---------------------------------------------------------------------------
__global__ __launch_bounds__(256)
void colstats_kernel(const float* __restrict__ L, float* __restrict__ stats, int Ntok, int NS) {
    int f = blockIdx.x;
    int tid = threadIdx.x;
    float s = 0.f, sq = 0.f;
    for (int n = tid; n < Ntok; n += 256) {
        float v = L[(size_t)n * NS + f];
        s += v;
        sq += v * v;
    }
    __shared__ float ss[256], sqs[256];
    ss[tid] = s; sqs[tid] = sq;
    __syncthreads();
    for (int st = 128; st > 0; st >>= 1) {
        if (tid < st) { ss[tid] += ss[tid + st]; sqs[tid] += sqs[tid + st]; }
        __syncthreads();
    }
    if (tid == 0) {
        float mean = ss[0] / (float)Ntok;
        float var  = sqs[0] / (float)Ntok - mean * mean;
        stats[f]      = mean;
        stats[NS + f] = rsqrtf(var + 1e-5f);
    }
}

// ---------------------------------------------------------------------------
// In-place batchnorm + row log_softmax over [Ntok, 128]; block = 128 = one row
// ---------------------------------------------------------------------------
__global__ __launch_bounds__(128)
void rowlsm_kernel(float* __restrict__ L, const float* __restrict__ stats, int NS) {
    int t = blockIdx.x, f = threadIdx.x;
    float v = (L[(size_t)t * NS + f] - stats[f]) * stats[NS + f];
    __shared__ float red[128];
    red[f] = v;
    __syncthreads();
    for (int s = 64; s > 0; s >>= 1) {
        if (f < s) red[f] = fmaxf(red[f], red[f + s]);
        __syncthreads();
    }
    float mx = red[0];
    __syncthreads();
    red[f] = __expf(v - mx);
    __syncthreads();
    for (int s = 64; s > 0; s >>= 1) {
        if (f < s) red[f] += red[f + s];
        __syncthreads();
    }
    float lse = mx + __logf(red[0]);
    L[(size_t)t * NS + f] = v - lse;
}

// ---------------------------------------------------------------------------
// Product-key top-16: top16(lpx) x top16(lpy) -> top16 of 256 candidate sums.
// One wave32 per token, 8 waves per block.
// ---------------------------------------------------------------------------
__global__ __launch_bounds__(256)
void topk_kernel(const float* __restrict__ LX, const float* __restrict__ LY,
                 int* __restrict__ tki, float* __restrict__ tkw, int Ntok) {
    int wv = threadIdx.x >> 5, lane = threadIdx.x & 31;
    int t = blockIdx.x * 8 + wv;
    if (t >= Ntok) return;
    const float* lx = LX + (size_t)t * 128;
    const float* ly = LY + (size_t)t * 128;

    float tv[16]; int ti[16];   // top-16 of lpx (replicated across lanes)
    float uv[16]; int ui[16];   // top-16 of lpy
    {
        float vx[4]; int ix[4];
#pragma unroll
        for (int j = 0; j < 4; ++j) { ix[j] = lane + 32 * j; vx[j] = lx[ix[j]]; }
        for (int r = 0; r < 16; ++r) {
            float bv = vx[0]; int bi = ix[0];
#pragma unroll
            for (int j = 1; j < 4; ++j)
                if (vx[j] > bv || (vx[j] == bv && ix[j] < bi)) { bv = vx[j]; bi = ix[j]; }
            for (int off = 16; off; off >>= 1) {
                float ov = __shfl_xor(bv, off, 32);
                int   oi = __shfl_xor(bi, off, 32);
                if (ov > bv || (ov == bv && oi < bi)) { bv = ov; bi = oi; }
            }
            tv[r] = bv; ti[r] = bi;
#pragma unroll
            for (int j = 0; j < 4; ++j)
                if (ix[j] == bi) vx[j] = NEG_INF;
        }
        float vy[4]; int iy[4];
#pragma unroll
        for (int j = 0; j < 4; ++j) { iy[j] = lane + 32 * j; vy[j] = ly[iy[j]]; }
        for (int r = 0; r < 16; ++r) {
            float bv = vy[0]; int bi = iy[0];
#pragma unroll
            for (int j = 1; j < 4; ++j)
                if (vy[j] > bv || (vy[j] == bv && iy[j] < bi)) { bv = vy[j]; bi = iy[j]; }
            for (int off = 16; off; off >>= 1) {
                float ov = __shfl_xor(bv, off, 32);
                int   oi = __shfl_xor(bi, off, 32);
                if (ov > bv || (ov == bv && oi < bi)) { bv = ov; bi = oi; }
            }
            uv[r] = bv; ui[r] = bi;
#pragma unroll
            for (int j = 0; j < 4; ++j)
                if (iy[j] == bi) vy[j] = NEG_INF;
        }
    }
    // 256 candidate sums, select 16 greatest
    float cv[8]; int ci[8];
#pragma unroll
    for (int q = 0; q < 8; ++q) {
        int p = lane + 32 * q;
        int a = p >> 4, b = p & 15;
        cv[q] = tv[a] + uv[b];
        ci[q] = ti[a] * 128 + ui[b];
    }
    for (int r = 0; r < 16; ++r) {
        float bv = cv[0]; int bi = ci[0];
#pragma unroll
        for (int q = 1; q < 8; ++q)
            if (cv[q] > bv || (cv[q] == bv && ci[q] < bi)) { bv = cv[q]; bi = ci[q]; }
        for (int off = 16; off; off >>= 1) {
            float ov = __shfl_xor(bv, off, 32);
            int   oi = __shfl_xor(bi, off, 32);
            if (ov > bv || (ov == bv && oi < bi)) { bv = ov; bi = oi; }
        }
        if (lane == 0) {
            tki[(size_t)t * 16 + r] = bi;
            tkw[(size_t)t * 16 + r] = __expf(bv);
        }
#pragma unroll
        for (int q = 0; q < 8; ++q)
            if (ci[q] == bi) cv[q] = NEG_INF;
    }
}

// ---------------------------------------------------------------------------
// PEER expert gather: per token block; 16 gathered dot products + rank-1 accum
// ---------------------------------------------------------------------------
__global__ __launch_bounds__(256)
void expert_kernel(const float* __restrict__ X, const float* __restrict__ UE,
                   const float* __restrict__ DE, const int* __restrict__ tki,
                   const float* __restrict__ tkw, float* __restrict__ expacc) {
    int t = blockIdx.x, tid = threadIdx.x;
    __shared__ float xs[1024];
    __shared__ float wred[8];
    __shared__ float ewsh;
    for (int i = tid; i < 1024; i += 256) xs[i] = X[(size_t)t * 1024 + i];
    __syncthreads();

    float acc[4] = {0.f, 0.f, 0.f, 0.f};
    for (int k = 0; k < 16; ++k) {
        int e = tki[(size_t)t * 16 + k];
        const float* ue = UE + (size_t)e * 1024;
        float part = 0.f;
#pragma unroll
        for (int j = 0; j < 4; ++j) {
            int i = tid + 256 * j;
            part += ue[i] * xs[i];
        }
        for (int off = 16; off; off >>= 1) part += __shfl_xor(part, off, 32);
        if ((tid & 31) == 0) wred[tid >> 5] = part;
        __syncthreads();
        if (tid == 0) {
            float d = 0.f;
#pragma unroll
            for (int w = 0; w < 8; ++w) d += wred[w];
            ewsh = silu_f(d) * tkw[(size_t)t * 16 + k];
        }
        __syncthreads();
        float ew = ewsh;
        const float* de = DE + (size_t)e * 1024;
#pragma unroll
        for (int j = 0; j < 4; ++j) {
            int i = tid + 256 * j;
            acc[j] += ew * de[i];
        }
        __syncthreads();
    }
#pragma unroll
    for (int j = 0; j < 4; ++j) {
        int i = tid + 256 * j;
        expacc[(size_t)t * 1024 + i] = acc[j];
    }
}

// ---------------------------------------------------------------------------
// Launcher
// ---------------------------------------------------------------------------
extern "C" void kernel_launch(void* const* d_in, const int* in_sizes, int n_in,
                              void* d_out, int out_size, void* d_ws, size_t ws_size,
                              hipStream_t stream) {
    const int H = 1024;
    const int Ntok = in_sizes[0] / H;       // 4096
    const int I    = in_sizes[1] / H;       // 4096
    const int NS   = in_sizes[4] / H;       // 128
    (void)n_in; (void)out_size; (void)ws_size;

    const float* x      = (const float*)d_in[0];
    const float* gate_w = (const float*)d_in[1];
    const float* up_w   = (const float*)d_in[2];
    const float* down_w = (const float*)d_in[3];
    const float* rxw    = (const float*)d_in[4];
    const float* ryw    = (const float*)d_in[5];
    const float* ue     = (const float*)d_in[6];
    const float* de     = (const float*)d_in[7];
    float* out = (float*)d_out;

    // workspace carve-up (256B aligned)
    char* p = (char*)d_ws;
    auto alloc = [&](size_t bytes) -> void* {
        void* r = (void*)p;
        p += (bytes + 255) & ~(size_t)255;
        return r;
    };
    unsigned short* xb   = (unsigned short*)alloc((size_t)Ntok * H * 2);
    unsigned short* gwb  = (unsigned short*)alloc((size_t)I * H * 2);
    unsigned short* uwb  = (unsigned short*)alloc((size_t)I * H * 2);
    unsigned short* dwb  = (unsigned short*)alloc((size_t)H * I * 2);
    unsigned short* rxb  = (unsigned short*)alloc((size_t)NS * H * 2);
    unsigned short* ryb  = (unsigned short*)alloc((size_t)NS * H * 2);
    float* lx    = (float*)alloc((size_t)Ntok * NS * 4);
    float* ly    = (float*)alloc((size_t)Ntok * NS * 4);
    float* statx = (float*)alloc((size_t)2 * NS * 4);
    float* staty = (float*)alloc((size_t)2 * NS * 4);
    int*   tki   = (int*)alloc((size_t)Ntok * 16 * 4);
    float* tkw   = (float*)alloc((size_t)Ntok * 16 * 4);
    float* eacc  = (float*)alloc((size_t)Ntok * H * 4);
    unsigned short* hbuf = (unsigned short*)alloc((size_t)Ntok * I * 2);

    // 1) f32 -> bf16 operand conversion
    cvt_bf16_kernel<<<2048, 256, 0, stream>>>(x,      xb,  Ntok * H);
    cvt_bf16_kernel<<<2048, 256, 0, stream>>>(gate_w, gwb, I * H);
    cvt_bf16_kernel<<<2048, 256, 0, stream>>>(up_w,   uwb, I * H);
    cvt_bf16_kernel<<<2048, 256, 0, stream>>>(down_w, dwb, H * I);
    cvt_bf16_kernel<<<256,  256, 0, stream>>>(rxw,    rxb, NS * H);
    cvt_bf16_kernel<<<256,  256, 0, stream>>>(ryw,    ryb, NS * H);

    // 2) router logits via WMMA GEMM  [Ntok,128] = x @ rw^T
    {
        dim3 grid(Ntok / 64, NS / 128);
        gemm_bf16_kernel<0><<<grid, 256, 0, stream>>>(xb, rxb, nullptr, nullptr, lx, nullptr, Ntok, NS, H);
        gemm_bf16_kernel<0><<<grid, 256, 0, stream>>>(xb, ryb, nullptr, nullptr, ly, nullptr, Ntok, NS, H);
    }
    // 3) per-feature batch stats; 4) batchnorm + log_softmax in place
    colstats_kernel<<<NS, 256, 0, stream>>>(lx, statx, Ntok, NS);
    colstats_kernel<<<NS, 256, 0, stream>>>(ly, staty, Ntok, NS);
    rowlsm_kernel<<<Ntok, 128, 0, stream>>>(lx, statx, NS);
    rowlsm_kernel<<<Ntok, 128, 0, stream>>>(ly, staty, NS);

    // 5) decomposed global top-16 over the 128x128 outer-sum grid
    topk_kernel<<<Ntok / 8, 256, 0, stream>>>(lx, ly, tki, tkw, Ntok);

    // 6) expert embedding gather + silu-weighted rank-1 accumulation
    expert_kernel<<<Ntok, 256, 0, stream>>>(x, ue, de, tki, tkw, eacc);

    // 7) dense gated MLP: fused gate/up WMMA GEMM -> h = silu(g)*u (bf16)
    {
        dim3 grid(Ntok / 64, I / 128);
        gemm_bf16_kernel<1><<<grid, 256, 0, stream>>>(xb, gwb, uwb, nullptr, nullptr, hbuf, Ntok, I, H);
    }
    // 8) down projection + expert residual -> d_out
    {
        dim3 grid(Ntok / 64, H / 128);
        gemm_bf16_kernel<2><<<grid, 256, 0, stream>>>(hbuf, dwb, nullptr, eacc, out, nullptr, Ntok, H, I);
    }
}
